// StyledConv_3917010174309
// MI455X (gfx1250) — compile-verified, matrix operand order checked by reading, not run on hardware
//
#include <hip/hip_runtime.h>

#define B_    16
#define CIN_  512
#define COUT_ 512
#define SD_   512
#define HW_   1024     // 32*32
#define PH_   34
#define PW_   36

#define CONV_SCALE  0.014731391274719738f   // 1/sqrt(512*9)
#define CONV_SCALE2 2.170138888888889e-4f   // 1/(512*9)
#define MOD_SCALE   0.044194173824159216f   // 1/sqrt(512)

typedef __bf16 bf16x16 __attribute__((ext_vector_type(16)));
typedef float  f32x8   __attribute__((ext_vector_type(8)));
typedef unsigned int u32x4 __attribute__((ext_vector_type(4)));
typedef int i32x4 __attribute__((ext_vector_type(4)));
typedef int i32x8 __attribute__((ext_vector_type(8)));

union FragU { uint4 q[2]; bf16x16 v; };

// LDS row stride for 32-wide K slabs, padded 32 -> 40 bf16 (64B -> 80B) by the TDM
#define LDSW 40
#define ASUP_ELEMS (6 * 36 * LDSW)          // 8640  (17280 B)
#define BSUP_ELEMS (8 * 9 * 16 * LDSW)      // 46080 (92160 B)
#define SMEM_BYTES ((ASUP_ELEMS + BSUP_ELEMS) * 2)

static __device__ __forceinline__ unsigned short f2bf(float f) {
    unsigned int u = __float_as_uint(f);
    unsigned int r = (u + 0x7FFFu + ((u >> 16) & 1u)) >> 16;   // RNE
    return (unsigned short)r;
}

// ---- Tensor Data Mover: DMA a (tile0 x tile1 x tile2) tile of 2-byte elements
// into LDS, padding every 64B row (32 elems) with 16B -> 80B LDS row stride.
// Strides/dims in elements. Per D# spec: Maddr = ga + 2*(Y*s0 + Z*s1).
static __device__ __forceinline__ void tdm_load_tile(
    unsigned lds_addr, unsigned long long ga,
    unsigned td0, unsigned td1, unsigned td2,
    unsigned t0, unsigned t1, unsigned t2,
    unsigned long long s0, unsigned long long s1)
{
    u32x4 g0;
    g0[0] = 1u;                                     // count=1, user descriptor
    g0[1] = lds_addr;                               // LDS byte address
    g0[2] = (unsigned)(ga & 0xFFFFFFFFull);         // global_addr[31:0]
    g0[3] = (unsigned)((ga >> 32) & 0x01FFFFFFull)  // global_addr[56:32]
          | 0x80000000u;                            // type=2 ("image") in [127:126]
    i32x8 g1;
    g1[0] = (int)((1u << 16)      // data_size = 2 bytes
                | (1u << 20)      // pad_enable
                | (3u << 22)      // pad_interval: every 16 DWORDs (64B)
                | (3u << 25));    // pad_amount: 4 DWORDs (16B)
    g1[1] = (int)((td0 & 0xFFFFu) << 16);                               // tensor_dim0 lo
    g1[2] = (int)(((td0 >> 16) & 0xFFFFu) | ((td1 & 0xFFFFu) << 16));   // dim0 hi | dim1 lo
    g1[3] = (int)(((td1 >> 16) & 0xFFFFu) | (t0 << 16));                // dim1 hi | tile_dim0
    g1[4] = (int)(t1 | (t2 << 16));                                     // tile_dim1 | tile_dim2
    g1[5] = (int)(unsigned)(s0 & 0xFFFFFFFFull);                        // dim0_stride lo32
    g1[6] = (int)((((unsigned)(s0 >> 32)) & 0xFFFFu)
                | (((unsigned)(s1 & 0xFFFFull)) << 16));                // s0 hi16 | s1 lo16
    g1[7] = (int)(unsigned)((s1 >> 16) & 0xFFFFFFFFull);                // s1 [47:16]
    i32x4 g2;
    g2[0] = (int)td2;  g2[1] = 0;  g2[2] = 0;  g2[3] = 0;               // tensor_dim2
    i32x4 g3;
    g3[0] = 0; g3[1] = 0; g3[2] = 0; g3[3] = 0;
#if defined(__clang_major__) && __clang_major__ >= 23
    i32x8 g4;
    g4[0]=0; g4[1]=0; g4[2]=0; g4[3]=0; g4[4]=0; g4[5]=0; g4[6]=0; g4[7]=0;
    __builtin_amdgcn_tensor_load_to_lds(g0, g1, g2, g3, g4, 0);
#else
    __builtin_amdgcn_tensor_load_to_lds(g0, g1, g2, g3, 0);
#endif
}

// ---- prep 1: s = style @ (mod_w^T * mod_scale) + mod_b ;  s2 = s*s ----
__global__ void prep_style(const float* __restrict__ style, const float* __restrict__ mod_w,
                           const float* __restrict__ mod_b, float* __restrict__ s,
                           float* __restrict__ s2) {
    int i = blockIdx.x * blockDim.x + threadIdx.x;      // B*CIN
    if (i >= B_ * CIN_) return;
    int b = i / CIN_, c = i % CIN_;
    const float* st = style + b * SD_;
    const float* mw = mod_w + c * SD_;
    float acc = 0.f;
    for (int k = 0; k < SD_; ++k) acc += st[k] * mw[k];
    float sv = acc * MOD_SCALE + mod_b[c];
    s[i] = sv;
    s2[i] = sv * sv;
}

// ---- prep 2: wsq[o,c] = sum_kk w^2 ; wtr[kk][o][c] = bf16(w) ----
__global__ void prep_weight(const float* __restrict__ weight, float* __restrict__ wsq,
                            unsigned short* __restrict__ wtr) {
    int i = blockIdx.x * blockDim.x + threadIdx.x;      // COUT*CIN
    if (i >= COUT_ * CIN_) return;
    const float* wp = weight + (size_t)i * 9;
    float ss = 0.f;
#pragma unroll
    for (int t = 0; t < 9; ++t) {
        float v = wp[t];
        ss += v * v;
        wtr[(size_t)t * (COUT_ * CIN_) + i] = f2bf(v);
    }
    wsq[i] = ss;
}

// ---- prep 3: demod[b,o] = rsqrt(conv_scale^2 * sum_c wsq[o,c]*s2[b,c] + 1e-8) ----
__global__ void prep_demod(const float* __restrict__ wsq, const float* __restrict__ s2,
                           float* __restrict__ demod) {
    int i = blockIdx.x * blockDim.x + threadIdx.x;      // B*COUT
    if (i >= B_ * COUT_) return;
    int b = i / COUT_, o = i % COUT_;
    const float* wq = wsq + o * CIN_;
    const float* sq = s2 + b * CIN_;
    float acc = 0.f;
    for (int c = 0; c < CIN_; ++c) acc += wq[c] * sq[c];
    demod[i] = rsqrtf(acc * CONV_SCALE2 + 1e-8f);
}

// ---- prep 4: xt[b][ph][pw][c] = bf16( x * s[b,c] * conv_scale ), zero border.
// Channel-last layout (K-contiguous) via an LDS 32x32 transpose so both global
// reads (along w) and writes (along c) stay coalesced.
// grid: (34 ph, 16 c-chunks, 16 b), 256 threads
__global__ __launch_bounds__(256)
void prep_xt(const float* __restrict__ x, const float* __restrict__ s,
             unsigned short* __restrict__ xt) {
    const int ph = blockIdx.x;
    const int c0 = blockIdx.y * 32;
    const int b  = blockIdx.z;
    const int tid = threadIdx.x;
    unsigned short* dst = xt + ((size_t)(b * PH_ + ph)) * PW_ * CIN_;   // + pw*512 + c

    if (ph == 0 || ph == 33) {          // top/bottom zero rows
        for (int j = tid; j < PW_ * 32; j += 256)
            dst[(j >> 5) * CIN_ + c0 + (j & 31)] = 0;
        return;
    }
    __shared__ unsigned short tile[32][34];   // [w][c], padded stride
    const int h = ph - 1;
#pragma unroll
    for (int cc = 0; cc < 4; ++cc) {
        int cl = (tid >> 5) + cc * 8;         // 0..31
        int w  = tid & 31;                    // coalesced read along w
        float v = x[(((size_t)(b * CIN_ + c0 + cl)) << 10) + h * 32 + w]
                * s[b * CIN_ + c0 + cl] * CONV_SCALE;
        tile[w][cl] = f2bf(v);
    }
    __syncthreads();
#pragma unroll
    for (int cc = 0; cc < 4; ++cc) {
        int w  = tid >> 3;                    // 0..31
        int cl = (tid & 7) + cc * 8;          // coalesced-ish write along c
        dst[(w + 1) * CIN_ + c0 + cl] = tile[w][cl];
    }
    if (tid < 128) {                          // side borders pw = 0,33,34,35
        int j = tid >> 5;
        int pw = (j == 0) ? 0 : 32 + j;
        dst[pw * CIN_ + c0 + (tid & 31)] = 0;
    }
}

// ---- main: implicit-GEMM conv, TDM-fed LDS, bf16 WMMA, fused epilogue ----
// Block tile: 128 positions (M, 4 image rows) x 128 couts (N). 8 waves: 4Mx2N.
// Outer loop: cin chunks of 32. TDM loads A supertile [6 rows][36 cols][32 ch]
// (shared by all 9 taps) and B [9 taps][16 couts][32 ch] per wave. Inner loop:
// 9 taps x 8 wmma; all LDS fragment addresses are base + compile-time offsets.
__global__ __launch_bounds__(256)
void styled_conv_wmma(const unsigned short* __restrict__ xt,
                      const unsigned short* __restrict__ wtr,
                      const float* __restrict__ demod,
                      const float* __restrict__ noise,
                      const float* __restrict__ nw,
                      const float* __restrict__ act_bias,
                      float* __restrict__ out) {
    extern __shared__ __align__(16) unsigned short smem[];
    unsigned short* Asup = smem;                 // [6 r][36 wp][40 k]
    unsigned short* Bsup = smem + ASUP_ELEMS;    // [8 wv][9 kk][16 n][40 k]

    const int tid  = threadIdx.x;
    const int lane = tid & 31;
    const int wv   = tid >> 5;
    const int waveM = wv & 3;     // M offset waveM*32
    const int waveN = wv >> 2;    // N offset waveN*64

    const int m0 = blockIdx.x * 128;
    const int n0 = blockIdx.y * 128;
    const int b  = m0 >> 10;
    const int h0 = (m0 & 1023) >> 5;            // 4 output rows h0..h0+3

    f32x8 acc[2][4];
#pragma unroll
    for (int mt = 0; mt < 2; ++mt)
#pragma unroll
        for (int nt = 0; nt < 4; ++nt)
            acc[mt][nt] = (f32x8){0.f, 0.f, 0.f, 0.f, 0.f, 0.f, 0.f, 0.f};

    // wave-uniform TDM setup (LDS aperture: low 32 bits of flat addr = LDS byte addr)
    const unsigned ldsA = (unsigned)(unsigned long long)(void*)Asup;
    const unsigned ldsB = (unsigned)(unsigned long long)(void*)Bsup;
    const int arow = (wv < 6) ? wv : wv - 6;    // waves 6,7 duplicate rows 0,1 (idempotent)
    const unsigned short* agbase =
        xt + ((size_t)(b * PH_ + h0 + arow)) * PW_ * CIN_;        // row (h0+arow), pw=0, c=0
    const unsigned short* bgbase =
        wtr + ((size_t)(n0 + wv * 16)) * CIN_;                     // kk=0 plane, n rows

    const int kbase = (lane >> 4) * 8;          // CDNA5 16-bit A/B fragment K base

    // per-lane LDS fragment base pointers: invariant across c0 and kk
    const unsigned short* aptr[2];
#pragma unroll
    for (int mt = 0; mt < 2; ++mt) {
        int ml = waveM * 32 + mt * 16 + (lane & 15);   // local position 0..127
        aptr[mt] = &Asup[((ml >> 5) * 36 + (ml & 31)) * LDSW + kbase];
    }
    const unsigned short* bptr[4];
#pragma unroll
    for (int nt = 0; nt < 4; ++nt) {
        int n = waveN * 64 + nt * 16 + (lane & 15);    // local cout 0..127
        bptr[nt] = &Bsup[((n >> 4) * (9 * 16) + (n & 15)) * LDSW + kbase];
    }

    for (int c0 = 0; c0 < CIN_; c0 += 32) {
        // A: 36 wp-rows x 32 ch for this wave's padded image row (shared by all taps)
        tdm_load_tile(ldsA + (unsigned)(arow * 36 * 80),
                      (unsigned long long)(agbase + c0),
                      /*td*/ CIN_, 1224, 1,
                      /*tile*/ 32, 36, 1,
                      /*strides*/ CIN_, PW_ * CIN_);
        // B: [9 kk][16 n][32 ch] for this wave's 16 couts
        tdm_load_tile(ldsB + (unsigned)(wv * 9 * 16 * 80),
                      (unsigned long long)(bgbase + c0),
                      /*td*/ CIN_, COUT_, 9,
                      /*tile*/ 32, 16, 9,
                      /*strides*/ CIN_, COUT_ * CIN_);
        __builtin_amdgcn_s_wait_tensorcnt(0);
        __syncthreads();

#pragma unroll
        for (int kk = 0; kk < 9; ++kk) {
            const int aoff = ((kk / 3) * 36 + (kk % 3)) * LDSW;   // tap shift, const
            const int boff = kk * 16 * LDSW;                      // tap plane, const
            FragU afrag[2], bfrag[4];
#pragma unroll
            for (int mt = 0; mt < 2; ++mt) {
                afrag[mt].q[0] = *(const uint4*)(aptr[mt] + aoff);        // K kb..kb+7
                afrag[mt].q[1] = *(const uint4*)(aptr[mt] + aoff + 16);   // K kb+16..kb+23
            }
#pragma unroll
            for (int nt = 0; nt < 4; ++nt) {
                bfrag[nt].q[0] = *(const uint4*)(bptr[nt] + boff);
                bfrag[nt].q[1] = *(const uint4*)(bptr[nt] + boff + 16);
            }
#pragma unroll
            for (int mt = 0; mt < 2; ++mt)
#pragma unroll
                for (int nt = 0; nt < 4; ++nt)
                    acc[mt][nt] = __builtin_amdgcn_wmma_f32_16x16x32_bf16(
                        false, afrag[mt].v, false, bfrag[nt].v,
                        (short)0, acc[mt][nt], false, false);
        }
        __syncthreads();
    }

    // ---- epilogue: demod, noise, bias, leaky_relu(0.2)*sqrt(2) ----
    const float nwv = nw[0];
    const float sq2 = 1.4142135623730951f;
    float dm[4], ab[4];
    int coutv[4];
#pragma unroll
    for (int nt = 0; nt < 4; ++nt) {
        coutv[nt] = n0 + waveN * 64 + nt * 16 + (lane & 15);
        dm[nt] = demod[b * COUT_ + coutv[nt]];
        ab[nt] = act_bias[coutv[nt]];
    }
    float* obase = out + ((size_t)b * COUT_ << 10);
#pragma unroll
    for (int mt = 0; mt < 2; ++mt) {
#pragma unroll
        for (int r = 0; r < 8; ++r) {
            const int mloc = waveM * 32 + mt * 16 + r + ((lane >> 4) << 3);
            const int pos  = (m0 + mloc) & 1023;
            const float nz = nwv * noise[b * HW_ + pos];
#pragma unroll
            for (int nt = 0; nt < 4; ++nt) {
                float v = acc[mt][nt][r] * dm[nt] + nz + ab[nt];
                v = (v > 0.f ? v : 0.2f * v) * sq2;
                obase[((size_t)coutv[nt] << 10) + pos] = v;
            }
        }
    }
}

extern "C" void kernel_launch(void* const* d_in, const int* in_sizes, int n_in,
                              void* d_out, int out_size, void* d_ws, size_t ws_size,
                              hipStream_t stream) {
    const float* x        = (const float*)d_in[0];
    const float* style    = (const float*)d_in[1];
    const float* noise    = (const float*)d_in[2];
    const float* weight   = (const float*)d_in[3];
    const float* mod_w    = (const float*)d_in[4];
    const float* mod_b    = (const float*)d_in[5];
    const float* noise_w  = (const float*)d_in[6];
    const float* act_bias = (const float*)d_in[7];
    float* out = (float*)d_out;

    char* ws = (char*)d_ws;
    float* s     = (float*)(ws + 0);                         // 16*512 f32
    float* s2    = (float*)(ws + 32768);                     // 16*512 f32
    float* demod = (float*)(ws + 65536);                     // 16*512 f32
    float* wsq   = (float*)(ws + 98304);                     // 512*512 f32
    unsigned short* wtr = (unsigned short*)(ws + 1146880);   // 9*512*512 bf16
    unsigned short* xt  = (unsigned short*)(ws + 5865472);   // 16*34*36*512 bf16

    prep_style<<<32, 256, 0, stream>>>(style, mod_w, mod_b, s, s2);
    prep_weight<<<1024, 256, 0, stream>>>(weight, wsq, wtr);
    prep_demod<<<32, 256, 0, stream>>>(wsq, s2, demod);
    prep_xt<<<dim3(PH_, 16, B_), 256, 0, stream>>>(x, s, xt);

    dim3 grid(128, 4);  // 16384/128 position tiles x 512/128 cout tiles
    styled_conv_wmma<<<grid, 256, SMEM_BYTES, stream>>>(xt, wtr, demod, noise,
                                                        noise_w, act_bias, out);
}
